// Fp16SegAttention_55843164782785
// MI455X (gfx1250) — compile-verified, matrix operand order checked by reading, not run on hardware
//
#include <hip/hip_runtime.h>
#include <cmath>

// ---- problem constants (match reference) ----
#define B_  4
#define Q_  1024
#define S_  2048
#define H_  32
#define D_  128
static constexpr float SCALE = 0.08838834764831845f;

typedef _Float16 h16;
typedef __attribute__((ext_vector_type(16))) _Float16 v16h;
typedef __attribute__((ext_vector_type(8)))  _Float16 v8h;
typedef __attribute__((ext_vector_type(4)))  _Float16 v4h;
typedef __attribute__((ext_vector_type(8)))  float    v8f;
typedef __attribute__((ext_vector_type(4)))  float    v4f;

#define KPAD 8   // halves of padding per K row (272B stride -> conflict-free)
#define VPAD 8   // VlT row = 40 halves = 80B stride
#define PPAD 8

__global__ __launch_bounds__(128, 1)
void seg_attn_kernel(const float* __restrict__ qs,
                     const float* __restrict__ kn,
                     const float* __restrict__ vn,
                     const float* __restrict__ kc,
                     const float* __restrict__ vc,
                     const int*   __restrict__ qlens,
                     const int*   __restrict__ clens,
                     float*       __restrict__ out)
{
    __shared__ __align__(16) h16 Kl[32][D_ + KPAD];        // K tile, [key][d]
    __shared__ __align__(16) h16 VlT[D_][32 + VPAD];       // V tile transposed, [d][key]
    __shared__ __align__(16) h16 Pl[4][16][32 + PPAD];     // per-wave probs, [row][key]

    const int qtiles = Q_ / 64;
    const int q0   = (blockIdx.x % qtiles) * 64;
    const int hb   = blockIdx.x / qtiles;
    const int h    = hb % H_;
    const int b    = hb / H_;
    const int tid  = threadIdx.x;
    const int wave = tid >> 5;
    const int lane = tid & 31;
    const int half = lane >> 4;      // 0: lanes 0-15, 1: lanes 16-31
    const int nn   = lane & 15;
    const int q0w  = q0 + wave * 16;

    const int qlen = qlens[b];
    const int clen = clens[b];

    // ---- preload Q fragments (A-matrix 16x32 f16 layout, 4 d-chunks) ----
    v16h qa[4];
    {
        const int row = q0w + nn;                 // A: lane%16 == M
        const float* qp = qs + (((size_t)b * Q_ + row) * H_ + h) * D_;
        #pragma unroll
        for (int dc = 0; dc < 4; ++dc) {
            const int d0 = dc * 32 + half * 8;    // lanes>=16 hold K=8..15 / 24..31
            v4f x0 = *(const v4f*)(qp + d0);
            v4f x1 = *(const v4f*)(qp + d0 + 4);
            v4f x2 = *(const v4f*)(qp + d0 + 16);
            v4f x3 = *(const v4f*)(qp + d0 + 20);
            v16h a;
            #pragma unroll
            for (int i = 0; i < 4; ++i) {
                a[i]      = (h16)x0[i];
                a[4 + i]  = (h16)x1[i];
                a[8 + i]  = (h16)x2[i];
                a[12 + i] = (h16)x3[i];
            }
            qa[dc] = a;
        }
    }

    // ---- accumulators ----
    v8f o_acc[8];
    #pragma unroll
    for (int dc = 0; dc < 8; ++dc) o_acc[dc] = (v8f){0.f,0.f,0.f,0.f,0.f,0.f,0.f,0.f};
    float m_run[8], l_run[8];
    #pragma unroll
    for (int r = 0; r < 8; ++r) { m_run[r] = -3.0e38f; l_run[r] = 0.0f; }

    int limit[8];                                // causal+segment bound per owned row
    #pragma unroll
    for (int r = 0; r < 8; ++r) {
        int row = q0w + r + 8 * half;            // C/D layout row
        int m   = row + 1; if (m > qlen) m = qlen;
        limit[r] = clen + m;
    }
    int mymax = q0w + 16; if (mymax > qlen) mymax = qlen;
    const int my_kmax = clen + mymax;
    int wgq = q0 + 64; if (wgq > qlen) wgq = qlen;
    const int wg_kmax = clen + wgq;
    const int ktiles  = (wg_kmax + 31) >> 5;

    const size_t kc_b = ((size_t)b * S_) * H_ * D_ + (size_t)h * D_;
    const size_t kn_b = ((size_t)b * Q_) * H_ * D_ + (size_t)h * D_;

    for (int kt = 0; kt < ktiles; ++kt) {
        const int kbase = kt * 32;
        __syncthreads();
        // ---- cooperative staging: K tile + V tile (transposed), cache/new select ----
        #pragma unroll
        for (int i = 0; i < 8; ++i) {
            const int f   = tid + 128 * i;        // 1024 float4 slots
            const int key = f >> 5;
            const int d   = (f & 31) << 2;
            const int j   = kbase + key;
            const float *sk, *sv;
            if (j < clen) {
                sk = kc + kc_b + (size_t)j * H_ * D_ + d;
                sv = vc + kc_b + (size_t)j * H_ * D_ + d;
            } else {
                int jn = j - clen; if (jn > Q_ - 1) jn = Q_ - 1;
                sk = kn + kn_b + (size_t)jn * H_ * D_ + d;
                sv = vn + kn_b + (size_t)jn * H_ * D_ + d;
            }
            if (kt + 1 < ktiles) __builtin_prefetch(sk + (size_t)32 * H_ * D_, 0, 1);
            v4f x = *(const v4f*)sk;
            v4h hx;
            #pragma unroll
            for (int c = 0; c < 4; ++c) hx[c] = (h16)x[c];
            *(v4h*)&Kl[key][d] = hx;
            v4f y = *(const v4f*)sv;
            #pragma unroll
            for (int c = 0; c < 4; ++c) VlT[d + c][key] = (h16)y[c];
        }
        __syncthreads();

        if (kbase < my_kmax) {
            // ---- S = Q * K^T : two 16x16 f32 tiles over 32 keys ----
            v8f c0 = (v8f){0.f,0.f,0.f,0.f,0.f,0.f,0.f,0.f};
            v8f c1 = c0;
            #pragma unroll
            for (int dc = 0; dc < 4; ++dc) {
                const int d0 = dc * 32 + half * 8;
                v8h lo0 = *(const v8h*)&Kl[nn][d0];
                v8h hi0 = *(const v8h*)&Kl[nn][d0 + 16];
                v8h lo1 = *(const v8h*)&Kl[16 + nn][d0];
                v8h hi1 = *(const v8h*)&Kl[16 + nn][d0 + 16];
                v16h b0 = __builtin_shufflevector(lo0, hi0, 0,1,2,3,4,5,6,7,8,9,10,11,12,13,14,15);
                v16h b1 = __builtin_shufflevector(lo1, hi1, 0,1,2,3,4,5,6,7,8,9,10,11,12,13,14,15);
                c0 = __builtin_amdgcn_wmma_f32_16x16x32_f16(false, qa[dc], false, b0, (short)0, c0, false, false);
                c1 = __builtin_amdgcn_wmma_f32_16x16x32_f16(false, qa[dc], false, b1, (short)0, c1, false, false);
            }

            // ---- online softmax over the 32-key tile ----
            float alpha[8];
            const int k0 = kbase + nn;
            const int k1 = kbase + 16 + nn;
            #pragma unroll
            for (int r = 0; r < 8; ++r) {
                float s0 = (k0 < limit[r]) ? c0[r] * SCALE : -3.0e38f;
                float s1 = (k1 < limit[r]) ? c1[r] * SCALE : -3.0e38f;
                float mx = fmaxf(s0, s1);
                #pragma unroll
                for (int mm = 8; mm; mm >>= 1) mx = fmaxf(mx, __shfl_xor(mx, mm, 32));
                const float mn = fmaxf(m_run[r], mx);
                const float a  = __expf(m_run[r] - mn);
                const float p0 = __expf(s0 - mn);
                const float p1 = __expf(s1 - mn);
                float sm = p0 + p1;
                #pragma unroll
                for (int mm = 8; mm; mm >>= 1) sm += __shfl_xor(sm, mm, 32);
                l_run[r] = l_run[r] * a + sm;
                m_run[r] = mn;
                alpha[r] = a;
                const int prow = r + 8 * half;
                Pl[wave][prow][nn]      = (h16)p0;
                Pl[wave][prow][16 + nn] = (h16)p1;
            }
            asm volatile("s_wait_dscnt 0x0" ::: "memory");   // P store -> P frag load (same wave)

            // ---- rescale O by alpha ----
            #pragma unroll
            for (int dc = 0; dc < 8; ++dc)
                #pragma unroll
                for (int r = 0; r < 8; ++r) o_acc[dc][r] *= alpha[r];

            // ---- O += P * V ----
            const int kk0 = half * 8;
            v8h plo = *(const v8h*)&Pl[wave][nn][kk0];
            v8h phi = *(const v8h*)&Pl[wave][nn][16 + kk0];
            v16h pf = __builtin_shufflevector(plo, phi, 0,1,2,3,4,5,6,7,8,9,10,11,12,13,14,15);
            #pragma unroll
            for (int dc = 0; dc < 8; ++dc) {
                const int n = dc * 16 + nn;
                v8h v0 = *(const v8h*)&VlT[n][kk0];
                v8h v1 = *(const v8h*)&VlT[n][16 + kk0];
                v16h vf = __builtin_shufflevector(v0, v1, 0,1,2,3,4,5,6,7,8,9,10,11,12,13,14,15);
                o_acc[dc] = __builtin_amdgcn_wmma_f32_16x16x32_f16(false, pf, false, vf, (short)0, o_acc[dc], false, false);
            }
        }
    }

    // ---- normalize + masked store ----
    #pragma unroll
    for (int dc = 0; dc < 8; ++dc) {
        const int d = dc * 16 + nn;
        #pragma unroll
        for (int r = 0; r < 8; ++r) {
            const int qi = q0w + r + 8 * half;
            float val = (qi < qlen) ? (o_acc[dc][r] / l_run[r]) : 0.0f;
            out[(((size_t)b * Q_ + qi) * H_ + h) * D_ + d] = val;
        }
    }
}

extern "C" void kernel_launch(void* const* d_in, const int* in_sizes, int n_in,
                              void* d_out, int out_size, void* d_ws, size_t ws_size,
                              hipStream_t stream) {
    const float* qs = (const float*)d_in[0];  // query_states  [B,Q,H,D]
    const float* kn = (const float*)d_in[1];  // key_states    [B,Q,H,D]
    const float* vn = (const float*)d_in[2];  // value_states  [B,Q,H,D]
    const float* kc = (const float*)d_in[3];  // k_cache       [B,S,H,D]
    const float* vc = (const float*)d_in[4];  // v_cache       [B,S,H,D]
    const int*   ql = (const int*)d_in[5];    // q_lens        [B]
    const int*   cl = (const int*)d_in[6];    // cache_lens    [B]
    float* out = (float*)d_out;               // [B,Q,H,D]

    dim3 grid(B_ * H_ * (Q_ / 64));
    dim3 block(128);
    seg_attn_kernel<<<grid, block, 0, stream>>>(qs, kn, vn, kc, vc, ql, cl, out);
}
